// udkws_aa1clap_29910152249996
// MI455X (gfx1250) — matrix operand order repaired and checked by reading, not compiled, not run on hardware
//
#include <hip/hip_runtime.h>
#include <hip/hip_bf16.h>
#include <math.h>

#define Bn   1024
#define Tt   100
#define DIN  384
#define DP   128
#define Ee   100
#define TP   112   // padded T/E rows (multiple of 16)

typedef __attribute__((ext_vector_type(16))) _Float16 v16h;
typedef __attribute__((ext_vector_type(8)))  float    v8f;
typedef _Float16 h16;

// ---------------------------------------------------------------------------
// WMMA operand loaders (wave32, v_wmma_f32_16x16x32_f16)
// A (16x32, MxK) per ISA: lane<16 -> M=lane, halves 0..7=K k0+0..7, 8..15=K k0+16..23
//                         lane>=16 -> M=lane-16, halves 0..7=K k0+8..15, 8..15=K k0+24..31
// B operand loaded from B^T storage (row n = column n of B): lane<16 -> N=lane,
//   halves 0..7=K k0+0..7, 8..15=K k0+16..23; lane>=16 -> N=lane-16, K k0+8..15, k0+24..31.
// ---------------------------------------------------------------------------
__device__ __forceinline__ v16h ld_atile(const h16* base, int ld, int row0, int k0) {
  int lane = threadIdx.x & 31;
  const h16* p = base + (size_t)(row0 + (lane & 15)) * ld + (k0 + ((lane >> 4) << 3));
  v16h r;
  ((uint4*)&r)[0] = *(const uint4*)(p);
  ((uint4*)&r)[1] = *(const uint4*)(p + 16);
  return r;
}

__device__ __forceinline__ v16h ld_btile(const h16* baseT, int ld, int n0, int k0) {
  int lane = threadIdx.x & 31;
  const h16* p = baseT + (size_t)(n0 + (lane & 15)) * ld + (k0 + ((lane >> 4) << 3));
  v16h r;
  ((uint4*)&r)[0] = *(const uint4*)(p);
  ((uint4*)&r)[1] = *(const uint4*)(p + 16);
  return r;
}

__device__ __forceinline__ v8f wmma_f16(v16h a, v16h b, v8f c) {
  return __builtin_amdgcn_wmma_f32_16x16x32_f16(false, a, false, b, (short)0, c, false, false);
}

// ---------------------------------------------------------------------------
// Kernel 1: pack weights to f16, transposed (B-operand layout), zero-padded
// ---------------------------------------------------------------------------
__global__ void k_prep(const float* an_w, const float* com_w,
                       const float* wk, const float* wv,
                       h16* wta, h16* wtc, h16* wkt, h16* wvt) {
  int i = blockIdx.x * blockDim.x + threadIdx.x;
  const int NW = DP * DIN;  // 49152
  const int NK = TP * DP;   // 14336
  if (i < NW) { int n = i / DIN, k = i % DIN; wta[i] = (h16)an_w[k * DP + n]; return; }
  i -= NW;
  if (i < NW) { int n = i / DIN, k = i % DIN; wtc[i] = (h16)com_w[k * DP + n]; return; }
  i -= NW;
  if (i < NK) { int n = i / DP, k = i % DP;
                wkt[i] = (n < Ee && k < Ee) ? (h16)wk[k * Ee + n] : (h16)0.f; return; }
  i -= NK;
  if (i < NK) { int n = i / DP, k = i % DP;
                wvt[i] = (n < Ee && k < Ee) ? (h16)wv[k * Ee + n] : (h16)0.f; return; }
}

// ---------------------------------------------------------------------------
// Kernel 2: LayerNorm over 384, write f16 rows
// ---------------------------------------------------------------------------
__global__ void k_ln(const float* an_x, const float* com_x,
                     const float* an_g, const float* an_b,
                     const float* com_g, const float* com_b, h16* xln) {
  int row = blockIdx.x;                  // 0 .. 2*B*T-1
  int s = row / (Bn * Tt);
  const float* src = (s ? com_x : an_x) + (size_t)(row % (Bn * Tt)) * DIN;
  const float* g  = s ? com_g : an_g;
  const float* bb = s ? com_b : an_b;
  __shared__ float red[128];
  int tid = threadIdx.x;
  float v0 = src[tid], v1 = src[tid + 128], v2 = src[tid + 256];
  red[tid] = v0 + v1 + v2; __syncthreads();
  for (int st = 64; st > 0; st >>= 1) { if (tid < st) red[tid] += red[tid + st]; __syncthreads(); }
  float mean = red[0] * (1.f / DIN); __syncthreads();
  float d0 = v0 - mean, d1 = v1 - mean, d2 = v2 - mean;
  red[tid] = d0 * d0 + d1 * d1 + d2 * d2; __syncthreads();
  for (int st = 64; st > 0; st >>= 1) { if (tid < st) red[tid] += red[tid + st]; __syncthreads(); }
  float inv = rsqrtf(red[0] * (1.f / DIN) + 1e-5f);
  h16* dst = xln + (size_t)row * DIN;
  dst[tid]       = (h16)(d0 * inv * g[tid]       + bb[tid]);
  dst[tid + 128] = (h16)(d1 * inv * g[tid + 128] + bb[tid + 128]);
  dst[tid + 256] = (h16)(d2 * inv * g[tid + 256] + bb[tid + 256]);
}

// ---------------------------------------------------------------------------
// Kernel 3: projection GEMM (M=B*T, N=128, K=384) + bias + SiLU + row-L2-norm
// 8 waves/block: one 16-row strip (LDS-staged), each wave = 16 output columns
// ---------------------------------------------------------------------------
__global__ void k_proj(const h16* xln, const h16* wta, const h16* wtc,
                       const float* an_b, const float* com_b, h16* P) {
  int bid = blockIdx.x;
  int s   = bid / (Bn * 7);
  int rem = bid % (Bn * 7);
  int b = rem / 7, mt = rem % 7;

  __shared__ alignas(16) h16 Astrip[16 * DIN];  // 12 KB
  __shared__ float rowsq[16];
  int tid = threadIdx.x;

  const h16* Xs = xln + (size_t)(s * Bn + b) * Tt * DIN;
  for (int i = tid; i < 16 * 192; i += 256) {   // 192 dwords per 384-half row
    int r = i / 192, c = i % 192;
    int t = mt * 16 + r;
    unsigned v = 0u;
    if (t < Tt) v = ((const unsigned*)(Xs + (size_t)t * DIN))[c];
    ((unsigned*)Astrip)[i] = v;
  }
  if (tid < 16) rowsq[tid] = 0.f;
  __syncthreads();

  int wave = tid >> 5, lane = tid & 31;
  int n0 = wave * 16;
  const h16* WT = s ? wtc : wta;

  v8f acc = {0.f, 0.f, 0.f, 0.f, 0.f, 0.f, 0.f, 0.f};
  for (int kt = 0; kt < DIN / 32; ++kt) {
    if (kt + 1 < DIN / 32)
      __builtin_prefetch(WT + (size_t)n0 * DIN + (kt + 1) * 32, 0, 1);
    v16h a  = ld_atile(Astrip, DIN, 0, kt * 32);
    v16h bt = ld_btile(WT, DIN, n0, kt * 32);
    acc = wmma_f16(a, bt, acc);
  }

  int n = lane & 15, mh = (lane >> 4) * 8;
  const float* bias = s ? com_b : an_b;
  float bvv = bias[n0 + n];
  float vals[8];
  for (int v = 0; v < 8; ++v) {
    float x = acc[v] + bvv;
    x = x / (1.f + __expf(-x));          // SiLU
    vals[v] = x;
    float ss = x * x;
    for (int m = 1; m < 16; m <<= 1) ss += __shfl_xor(ss, m, 32);
    if (n == 0) atomicAdd(&rowsq[v + mh], ss);
  }
  __syncthreads();

  h16* dst = P + (size_t)(s * Bn + b) * TP * DP;
  for (int v = 0; v < 8; ++v) {
    int row = mt * 16 + v + mh;
    float iv = 1.f / fmaxf(sqrtf(rowsq[v + mh]), 1e-8f);
    dst[(size_t)row * DP + n0 + n] = (h16)(vals[v] * iv);
  }
}

// ---------------------------------------------------------------------------
// Kernel 4: cosine similarity sim = an_n @ com_n^T, one wave per 16x16 tile
// ---------------------------------------------------------------------------
__global__ void k_sim(const h16* P, h16* sim) {
  int bid = blockIdx.x;
  int b = bid / 49, t = bid % 49;
  int mt = t / 7, nt = t % 7;
  const h16* Pa = P + (size_t)b * TP * DP;
  const h16* Pc = P + (size_t)(Bn + b) * TP * DP;
  v8f acc = {0.f, 0.f, 0.f, 0.f, 0.f, 0.f, 0.f, 0.f};
  for (int kt = 0; kt < DP / 32; ++kt) {
    v16h a  = ld_atile(Pa, DP, mt * 16, kt * 32);
    v16h bt = ld_btile(Pc, DP, nt * 16, kt * 32);
    acc = wmma_f16(a, bt, acc);
  }
  int lane = threadIdx.x & 31;
  int n = lane & 15, mh = (lane >> 4) * 8;
  int col = nt * 16 + n;
  h16* dst = sim + (size_t)b * Ee * Ee;
  for (int v = 0; v < 8; ++v) {
    int row = mt * 16 + v + mh;
    if (row < Ee && col < Ee) dst[row * Ee + col] = (h16)acc[v];
  }
}

// ---------------------------------------------------------------------------
// Kernel 5: masked crop + bilinear resize (align_corners) + 5x5 gaussian SAME
// writes zero-padded f16 smooth [112,128] and column-max q_in
// ---------------------------------------------------------------------------
__global__ void k_resize(const h16* sim, const int* amask, const int* cmask,
                         h16* smh, float* qin) {
  int b = blockIdx.x, tid = threadIdx.x;
  __shared__ float up[Ee * Ee];       // 40 KB
  __shared__ int   y0s[Ee], y1s[Ee], x0s[Ee], x1s[Ee];
  __shared__ float fys[Ee], fxs[Ee];
  __shared__ int   lsh[2];

  if (tid == 0) { int sl = 0; for (int t = 0; t < Tt; ++t) sl += amask[b * Tt + t]; lsh[0] = sl; }
  if (tid == 1) { int sl = 0; for (int t = 0; t < Tt; ++t) sl += cmask[b * Tt + t]; lsh[1] = sl; }
  __syncthreads();
  int la = lsh[0], lc = lsh[1];
  if (tid < Ee) {
    float sy = (la > 1) ? (float)(la - 1) / (float)(Ee - 1) : 0.f;
    float cy = tid * sy; int c0 = (int)floorf(cy);
    int c1 = c0 + 1; if (c1 > la - 1) c1 = la - 1;
    y0s[tid] = c0; y1s[tid] = c1; fys[tid] = cy - (float)c0;
    float sx = (lc > 1) ? (float)(lc - 1) / (float)(Ee - 1) : 0.f;
    float cx = tid * sx; int d0 = (int)floorf(cx);
    int d1 = d0 + 1; if (d1 > lc - 1) d1 = lc - 1;
    x0s[tid] = d0; x1s[tid] = d1; fxs[tid] = cx - (float)d0;
  }
  __syncthreads();

  const h16* sb = sim + (size_t)b * Ee * Ee;
  for (int idx = tid; idx < Ee * Ee; idx += blockDim.x) {
    int y = idx / Ee, x = idx % Ee;
    int r0 = y0s[y] * Ee, r1 = y1s[y] * Ee; float fy = fys[y];
    int c0 = x0s[x], c1 = x1s[x]; float fx = fxs[x];
    float v00 = (float)sb[r0 + c0], v01 = (float)sb[r0 + c1];
    float v10 = (float)sb[r1 + c0], v11 = (float)sb[r1 + c1];
    float ra = v00 + (v01 - v00) * fx;
    float rb = v10 + (v11 - v10) * fx;
    up[idx] = ra + (rb - ra) * fy;
  }
  __syncthreads();

  float gw[25]; float gs = 0.f;
  for (int dy = 0; dy < 5; ++dy)
    for (int dx = 0; dx < 5; ++dx) {
      float yy = (float)(dy - 2), xx = (float)(dx - 2);
      float w = __expf(-(yy * yy + xx * xx) * 0.5f);
      gw[dy * 5 + dx] = w; gs += w;
    }
  float gi = 1.f / gs;

  h16* so = smh + (size_t)b * TP * DP;
  for (int x = tid; x < Ee; x += blockDim.x) {
    float cmax = -1e30f;
    for (int y = 0; y < Ee; ++y) {
      float a = 0.f;
      for (int dy = -2; dy <= 2; ++dy) {
        int yy = y + dy; if (yy < 0 || yy >= Ee) continue;
        for (int dx = -2; dx <= 2; ++dx) {
          int xx = x + dx; if (xx < 0 || xx >= Ee) continue;
          a += up[yy * Ee + xx] * gw[(dy + 2) * 5 + dx + 2];
        }
      }
      a *= gi;
      so[(size_t)y * DP + x] = (h16)a;
      cmax = fmaxf(cmax, a);
    }
    qin[b * Ee + x] = cmax;
  }
  for (int i = tid; i < TP * DP; i += blockDim.x) {
    int y = i / DP, x = i % DP;
    if (y >= Ee || x >= Ee) so[i] = (h16)0.f;
  }
}

// ---------------------------------------------------------------------------
// Kernel 6: K = smooth@wk + bk, V = smooth@wv + bv  (wave0 -> K, wave1 -> V)
// ---------------------------------------------------------------------------
__global__ void k_kv(const h16* smh, const h16* wkt, const h16* wvt,
                     const float* bk, const float* bv, float* KB, float* VB) {
  int bid = blockIdx.x;
  int b = bid / 49, t = bid % 49;
  int mt = t / 7, nt = t % 7;
  int wave = threadIdx.x >> 5, lane = threadIdx.x & 31;
  const h16* A = smh + (size_t)b * TP * DP;
  const h16* W = wave ? wvt : wkt;
  const float* bias = wave ? bv : bk;
  float* O = (wave ? VB : KB) + (size_t)b * Ee * Ee;
  v8f acc = {0.f, 0.f, 0.f, 0.f, 0.f, 0.f, 0.f, 0.f};
  for (int kt = 0; kt < DP / 32; ++kt) {
    v16h a  = ld_atile(A, DP, mt * 16, kt * 32);
    v16h bt = ld_btile(W, DP, nt * 16, kt * 32);
    acc = wmma_f16(a, bt, acc);
  }
  int n = lane & 15, mh = (lane >> 4) * 8, col = nt * 16 + n;
  for (int v = 0; v < 8; ++v) {
    int row = mt * 16 + v + mh;
    if (row < Ee && col < Ee) O[row * Ee + col] = acc[v] + bias[col];
  }
}

// ---------------------------------------------------------------------------
// Kernel 7: attention head + FFN + dense (one 128-thread block per batch)
// ---------------------------------------------------------------------------
__device__ __forceinline__ float blk_sum(float v, float* red) {
  int tid = threadIdx.x;
  red[tid] = v; __syncthreads();
  for (int s = 64; s > 0; s >>= 1) { if (tid < s) red[tid] += red[tid + s]; __syncthreads(); }
  float r = red[0]; __syncthreads();
  return r;
}
__device__ __forceinline__ float blk_max(float v, float* red) {
  int tid = threadIdx.x;
  red[tid] = v; __syncthreads();
  for (int s = 64; s > 0; s >>= 1) { if (tid < s) red[tid] = fmaxf(red[tid], red[tid + s]); __syncthreads(); }
  float r = red[0]; __syncthreads();
  return r;
}

__global__ void k_attn(const float* qin, const float* KB, const float* VB,
                       const float* wq, const float* bq,
                       const float* wo, const float* bo,
                       const float* ln1g, const float* ln1b,
                       const float* ln2g, const float* ln2b,
                       const float* f1w, const float* f1b,
                       const float* f2w, const float* f2b,
                       const float* dw, const float* db, float* out) {
  int b = blockIdx.x, tid = threadIdx.x;
  __shared__ float q[Ee], Qv[Ee], sc[Ee], tmp[Ee], xv[Ee], red[128];

  if (tid < Ee) q[tid] = qin[b * Ee + tid];
  __syncthreads();
  if (tid < Ee) {
    float a = 0.f;
    for (int j = 0; j < Ee; ++j) a += q[j] * wq[j * Ee + tid];
    Qv[tid] = a + bq[tid];
  }
  __syncthreads();
  if (tid < Ee) {
    const float* Kr = KB + (size_t)b * Ee * Ee + tid * Ee;
    float a = 0.f;
    for (int e = 0; e < Ee; ++e) a += Qv[e] * Kr[e];
    sc[tid] = a * 0.1f;                       // / sqrt(100)
  }
  __syncthreads();
  float mx = blk_max(tid < Ee ? sc[tid] : -1e30f, red);
  float ex = tid < Ee ? __expf(sc[tid] - mx) : 0.f;
  float ssum = blk_sum(ex, red);
  if (tid < Ee) sc[tid] = ex / ssum;
  __syncthreads();
  if (tid < Ee) {
    const float* Vb = VB + (size_t)b * Ee * Ee;
    float a = 0.f;
    for (int k = 0; k < Ee; ++k) a += sc[k] * Vb[k * Ee + tid];
    tmp[tid] = a;
  }
  __syncthreads();
  if (tid < Ee) {
    float a = 0.f;
    for (int j = 0; j < Ee; ++j) a += tmp[j] * wo[j * Ee + tid];
    xv[tid] = q[tid] + a + bo[tid];
  }
  __syncthreads();
  // LN1
  float mean = blk_sum(tid < Ee ? xv[tid] : 0.f, red) * (1.f / Ee);
  float d = tid < Ee ? xv[tid] - mean : 0.f;
  float var = blk_sum(d * d, red) * (1.f / Ee);
  float inv = rsqrtf(var + 1e-5f);
  if (tid < Ee) xv[tid] = d * inv * ln1g[tid] + ln1b[tid];
  __syncthreads();
  // FFN
  if (tid < Ee) {
    float a = 0.f;
    for (int j = 0; j < Ee; ++j) a += xv[j] * f1w[j * Ee + tid];
    tmp[tid] = fmaxf(a + f1b[tid], 0.f);
  }
  __syncthreads();
  if (tid < Ee) {
    float a = 0.f;
    for (int j = 0; j < Ee; ++j) a += tmp[j] * f2w[j * Ee + tid];
    sc[tid] = xv[tid] + a + f2b[tid];
  }
  __syncthreads();
  // LN2 + dense
  mean = blk_sum(tid < Ee ? sc[tid] : 0.f, red) * (1.f / Ee);
  d = tid < Ee ? sc[tid] - mean : 0.f;
  var = blk_sum(d * d, red) * (1.f / Ee);
  inv = rsqrtf(var + 1e-5f);
  float contrib = tid < Ee ? (d * inv * ln2g[tid] + ln2b[tid]) * dw[tid] : 0.f;
  float tot = blk_sum(contrib, red);
  if (tid == 0) out[b] = tot + db[0];
}

// ---------------------------------------------------------------------------
// Launch: sequential pipeline on one stream; big LN buffer aliased by later
// intermediates (dead by then). Peak workspace ~216.3 MB.
// ---------------------------------------------------------------------------
extern "C" void kernel_launch(void* const* d_in, const int* in_sizes, int n_in,
                              void* d_out, int out_size, void* d_ws, size_t ws_size,
                              hipStream_t stream) {
  (void)in_sizes; (void)n_in; (void)out_size; (void)ws_size;
  const float* an_x  = (const float*)d_in[0];
  const float* com_x = (const float*)d_in[1];
  const int*   amask = (const int*)d_in[2];
  const int*   cmask = (const int*)d_in[3];
  const float* an_g  = (const float*)d_in[4];
  const float* an_lb = (const float*)d_in[5];
  const float* com_g = (const float*)d_in[6];
  const float* com_lb= (const float*)d_in[7];
  const float* an_w  = (const float*)d_in[8];
  const float* an_b  = (const float*)d_in[9];
  const float* com_w = (const float*)d_in[10];
  const float* com_b = (const float*)d_in[11];
  const float* wq = (const float*)d_in[12];
  const float* bq = (const float*)d_in[13];
  const float* wk = (const float*)d_in[14];
  const float* bk = (const float*)d_in[15];
  const float* wv = (const float*)d_in[16];
  const float* bv = (const float*)d_in[17];
  const float* wo = (const float*)d_in[18];
  const float* bo = (const float*)d_in[19];
  const float* ln1g = (const float*)d_in[20];
  const float* ln1b = (const float*)d_in[21];
  const float* ln2g = (const float*)d_in[22];
  const float* ln2b = (const float*)d_in[23];
  const float* f1w = (const float*)d_in[24];
  const float* f1b = (const float*)d_in[25];
  const float* f2w = (const float*)d_in[26];
  const float* f2b = (const float*)d_in[27];
  const float* dw  = (const float*)d_in[28];
  const float* db  = (const float*)d_in[29];
  float* out = (float*)d_out;

  char* w = (char*)d_ws;
  h16*   XLN = (h16*)w;                       // 157,286,400 B (dead after k_proj)
  h16*   SIM = (h16*)w;                       //  20,480,000 B (aliases XLN)
  h16*   SMH = (h16*)(w + 20480000);          //  29,360,128 B
  float* KB  = (float*)(w + 49840128);        //  40,960,000 B
  float* VB  = (float*)(w + 90800128);        //  40,960,000 B
  float* QIN = (float*)(w + 131760128);       //     409,600 B
  h16*   P   = (h16*)(w + 157286400);         //  58,720,256 B
  h16*   WTA = (h16*)(w + 216006656);         //      98,304 B
  h16*   WTC = (h16*)(w + 216104960);         //      98,304 B
  h16*   WKT = (h16*)(w + 216203264);         //      28,672 B
  h16*   WVT = (h16*)(w + 216231936);         //      28,672 B

  k_prep  <<<dim3(496),         256, 0, stream>>>(an_w, com_w, wk, wv, WTA, WTC, WKT, WVT);
  k_ln    <<<dim3(2 * Bn * Tt), 128, 0, stream>>>(an_x, com_x, an_g, an_lb, com_g, com_lb, XLN);
  k_proj  <<<dim3(2 * Bn * 7),  256, 0, stream>>>(XLN, WTA, WTC, an_b, com_b, P);
  k_sim   <<<dim3(Bn * 49),      32, 0, stream>>>(P, SIM);
  k_resize<<<dim3(Bn),          128, 0, stream>>>(SIM, amask, cmask, SMH, QIN);
  k_kv    <<<dim3(Bn * 49),      64, 0, stream>>>(SMH, WKT, WVT, bk, bv, KB, VB);
  k_attn  <<<dim3(Bn),          128, 0, stream>>>(QIN, KB, VB, wq, bq, wo, bo,
                                                  ln1g, ln1b, ln2g, ln2b,
                                                  f1w, f1b, f2w, f2b, dw, db, out);
}
